// GraphAttentionAggregator_6614249636059
// MI455X (gfx1250) — matrix operand order repaired and checked by reading, not compiled, not run on hardware
//
#include <hip/hip_runtime.h>

typedef __attribute__((ext_vector_type(16))) __bf16 v16bf;
typedef __attribute__((ext_vector_type(8)))  __bf16 v8bf;
typedef __attribute__((ext_vector_type(8)))  float  v8f;

// ---------- helpers ----------
__device__ __forceinline__ __bf16 f2bf(float f) {
  unsigned u = __float_as_uint(f);
  u += 0x7fffu + ((u >> 16) & 1u);                 // round to nearest even
  return __builtin_bit_cast(__bf16, (unsigned short)(u >> 16));
}

__device__ __forceinline__ float lrelu(float v) { return v > 0.f ? v : 0.2f * v; }

// order-preserving float<->uint encoding so atomicMax(uint) == float max
__device__ __forceinline__ unsigned fenc(float f) {
  unsigned u = __float_as_uint(f);
  return (u & 0x80000000u) ? ~u : (u | 0x80000000u);
}
__device__ __forceinline__ float fdec(unsigned o) {
  return (o & 0x80000000u) ? __uint_as_float(o & 0x7fffffffu) : __uint_as_float(~o);
}

// A-fragment (16x32 bf16, row-major source):
// lane L: row = L&15; q = L>>4; elems 0..15 -> K = kk+q*16+i
__device__ __forceinline__ v16bf load_afrag(const __bf16* rowbase, int kk, int q) {
  v8bf lo = *(const v8bf*)(rowbase + kk + q * 16);
  v8bf hi = *(const v8bf*)(rowbase + kk + q * 16 + 8);
  return __builtin_shufflevector(lo, hi, 0,1,2,3,4,5,6,7,8,9,10,11,12,13,14,15);
}

// ---------- kernel 0: transpose-convert weights to bf16 ----------
__global__ void k_wconv(const float* __restrict__ Wg, const float* __restrict__ W1,
                        const float* __restrict__ W2, __bf16* __restrict__ Wgt,
                        __bf16* __restrict__ W1t, __bf16* __restrict__ W2t) {
  int t = blockIdx.x * 256 + threadIdx.x;
  if (t < 128 * 128) {                        // W_gat [128,128] -> Wgt[n*128+k]
    int k = t >> 7, n = t & 127;
    Wgt[n * 128 + k] = f2bf(Wg[t]);
  } else if (t < 128 * 128 + 128 * 512) {     // W1 [128,512] -> W1t[n*128+k]
    int i = t - 128 * 128;
    int k = i >> 9, n = i & 511;
    W1t[n * 128 + k] = f2bf(W1[i]);
  } else if (t < 128 * 128 + 2 * 128 * 512) { // W2 [512,128] -> W2t[n*512+k]
    int i = t - 128 * 128 - 128 * 512;
    int k = i >> 7, n = i & 127;
    W2t[n * 512 + k] = f2bf(W2[i]);
  }
}

// ---------- kernel 1: init segment buffers ----------
__global__ void k_init(unsigned* __restrict__ segmax, float* __restrict__ denom,
                       float* __restrict__ agg, int Nn) {
  int t = blockIdx.x * 256 + threadIdx.x;
  int n8 = Nn * 8;
  if (t < n8) segmax[t] = fenc(-1e30f);
  else if (t < 2 * n8) denom[t - n8] = 0.f;
  else if (t < 2 * n8 + Nn * 128) agg[t - 2 * n8] = 0.f;
}

// ---------- kernel 2: h = x @ W_gat  (WMMA bf16, block = 16 rows x 128 cols) ----------
__global__ __launch_bounds__(256) void k_gemm_gat(const float* __restrict__ X,
                                                  const __bf16* __restrict__ Wt,
                                                  float* __restrict__ H, int Nn) {
  __shared__ __bf16 sA[16 * 128];   // 4 KB
  __shared__ float  sC[16 * 128];   // 8 KB (coalescing stage for C)
  int tid = threadIdx.x;
  int node0 = blockIdx.x * 16;
  bool full = (node0 + 16 <= Nn);   // uniform
  for (int i = tid; i < 16 * 128; i += 256) {
    int row = node0 + (i >> 7);
    if (row >= Nn) row = Nn - 1;
    sA[i] = f2bf(X[(size_t)row * 128 + (i & 127)]);
  }
  __syncthreads();
  int w = tid >> 5, lane = tid & 31, q = lane >> 4, mr = lane & 15;
  v8f acc = {};
  const __bf16* rowA = sA + mr * 128;
  const __bf16* Bt = Wt + (size_t)(w * 16 + mr) * 128;  // B col = lane&15, contiguous K
  for (int kk = 0; kk < 128; kk += 32) {
    v16bf a = load_afrag(rowA, kk, q);
    v16bf b = *(const v16bf*)(Bt + kk + q * 16);
    acc = __builtin_amdgcn_wmma_f32_16x16x32_bf16(false, a, false, b, (short)0, acc, false, false);
  }
  int col = w * 16 + mr;
  for (int r = 0; r < 8; ++r) sC[(q * 8 + r) * 128 + col] = acc[r];
  __syncthreads();
  if (full) {                       // coalesced b128 stores, uniform branch
    for (int i = tid * 8; i < 16 * 128; i += 256 * 8) {
      int row = node0 + (i >> 7);
      float* dst = H + (size_t)row * 128 + (i & 127);
      *(float4*)dst = *(const float4*)(sC + i);
      *(float4*)(dst + 4) = *(const float4*)(sC + i + 4);
    }
  } else {
    for (int i = tid; i < 16 * 128; i += 256) {
      int row = node0 + (i >> 7);
      if (row < Nn) H[(size_t)row * 128 + (i & 127)] = sC[i];
    }
  }
}

// ---------- kernel 3: per-(node,head) attention scores ----------
__global__ void k_att(const float* __restrict__ H, const float* __restrict__ asv,
                      const float* __restrict__ adv, float* __restrict__ a_src,
                      float* __restrict__ a_dst, int Nn) {
  int t = blockIdx.x * 256 + threadIdx.x;
  if (t >= Nn * 8) return;
  int n = t >> 3, hd = t & 7;
  const float* hp = H + (size_t)n * 128 + hd * 16;
  const float* sp = asv + hd * 16;
  const float* dp = adv + hd * 16;
  float ss = 0.f, sd = 0.f;
  for (int i = 0; i < 16; ++i) { float v = hp[i]; ss += v * sp[i]; sd += v * dp[i]; }
  a_src[t] = ss;
  a_dst[t] = sd;
}

// ---------- kernel 4: segment max over destinations ----------
__global__ void k_edge_max(const long long* __restrict__ ei, const float* __restrict__ a_src,
                           const float* __restrict__ a_dst, unsigned* __restrict__ segmax,
                           int Ee, int Nn) {
  int t = blockIdx.x * 256 + threadIdx.x;
  if (t >= Ee + Nn) return;
  long long s, d;
  if (t < Ee) { s = ei[t]; d = ei[(size_t)Ee + t]; } else { s = d = t - Ee; }
  for (int hd = 0; hd < 8; ++hd) {
    float el = lrelu(a_src[(size_t)s * 8 + hd] + a_dst[(size_t)d * 8 + hd]);
    atomicMax(&segmax[(size_t)d * 8 + hd], fenc(el));
  }
}

// ---------- kernel 5: segment sum of exp ----------
__global__ void k_edge_sum(const long long* __restrict__ ei, const float* __restrict__ a_src,
                           const float* __restrict__ a_dst, const unsigned* __restrict__ segmax,
                           float* __restrict__ denom, int Ee, int Nn) {
  int t = blockIdx.x * 256 + threadIdx.x;
  if (t >= Ee + Nn) return;
  long long s, d;
  if (t < Ee) { s = ei[t]; d = ei[(size_t)Ee + t]; } else { s = d = t - Ee; }
  for (int hd = 0; hd < 8; ++hd) {
    float el = lrelu(a_src[(size_t)s * 8 + hd] + a_dst[(size_t)d * 8 + hd]);
    float wv = expf(el - fdec(segmax[(size_t)d * 8 + hd]));
    atomicAdd(&denom[(size_t)d * 8 + hd], wv);
  }
}

// ---------- kernel 6: weighted message scatter (one thread per edge-head) ----------
__global__ void k_edge_scatter(const long long* __restrict__ ei, const float* __restrict__ a_src,
                               const float* __restrict__ a_dst, const unsigned* __restrict__ segmax,
                               const float* __restrict__ denom, const float* __restrict__ H,
                               float* __restrict__ agg, int Ee, int Nn) {
  int idx = blockIdx.x * 256 + threadIdx.x;
  if (idx >= (Ee + Nn) * 8) return;
  int t = idx >> 3, hd = idx & 7;
  long long s, d;
  if (t < Ee) { s = ei[t]; d = ei[(size_t)Ee + t]; } else { s = d = t - Ee; }
  float el = lrelu(a_src[(size_t)s * 8 + hd] + a_dst[(size_t)d * 8 + hd]);
  float p = expf(el - fdec(segmax[(size_t)d * 8 + hd]));
  float alpha = p / (denom[(size_t)d * 8 + hd] + 1e-16f);
  const float* hs = H + (size_t)s * 128 + hd * 16;
  float* ad = agg + (size_t)d * 128 + hd * 16;
  for (int i = 0; i < 16; ++i) atomicAdd(ad + i, hs[i] * alpha);
}

// ---------- kernel 7: h1 = LayerNorm(x + agg + b_gat)  (one wave per node) ----------
__global__ __launch_bounds__(256) void k_ln1(const float* __restrict__ X, const float* __restrict__ agg,
                                             const float* __restrict__ bg, const float* __restrict__ gamma,
                                             const float* __restrict__ beta, float* __restrict__ H1, int Nn) {
  int node = blockIdx.x * 8 + (threadIdx.x >> 5);
  int lane = threadIdx.x & 31;
  if (node >= Nn) return;
  float4 xv = *(const float4*)(X + (size_t)node * 128 + lane * 4);
  float4 av = *(const float4*)(agg + (size_t)node * 128 + lane * 4);
  float4 bv = *(const float4*)(bg + lane * 4);
  float4 v;
  v.x = xv.x + av.x + bv.x; v.y = xv.y + av.y + bv.y;
  v.z = xv.z + av.z + bv.z; v.w = xv.w + av.w + bv.w;
  float s = v.x + v.y + v.z + v.w;
  float qq = v.x * v.x + v.y * v.y + v.z * v.z + v.w * v.w;
  for (int off = 16; off; off >>= 1) {
    s += __shfl_xor(s, off, 32);
    qq += __shfl_xor(qq, off, 32);
  }
  float mu = s * (1.f / 128.f);
  float var = qq * (1.f / 128.f) - mu * mu;
  float ri = rsqrtf(var + 1e-5f);
  float4 g4 = *(const float4*)(gamma + lane * 4);
  float4 b4 = *(const float4*)(beta + lane * 4);
  float4 o;
  o.x = g4.x * (v.x - mu) * ri + b4.x; o.y = g4.y * (v.y - mu) * ri + b4.y;
  o.z = g4.z * (v.z - mu) * ri + b4.z; o.w = g4.w * (v.w - mu) * ri + b4.w;
  *(float4*)(H1 + (size_t)node * 128 + lane * 4) = o;
}

// ---------- kernel 8: fused FFN: gelu(h1@W1+b1)@W2+b2, residual, LayerNorm ----------
__global__ __launch_bounds__(256) void k_ffn(const float* __restrict__ H1, const __bf16* __restrict__ W1t,
                                             const float* __restrict__ b1, const __bf16* __restrict__ W2t,
                                             const float* __restrict__ b2, const float* __restrict__ gamma,
                                             const float* __restrict__ beta, float* __restrict__ OUT, int Nn) {
  __shared__ float  sH1f[16 * 128];   // 8 KB
  __shared__ __bf16 sH1b[16 * 128];   // 4 KB
  __shared__ __bf16 sMid[16 * 512];   // 16 KB (FFN intermediate never hits HBM)
  __shared__ float  sOut[16 * 128];   // 8 KB
  __shared__ float  sRedS[256], sRedQ[256];
  __shared__ float  sMu[16], sRi[16];
  int tid = threadIdx.x;
  int node0 = blockIdx.x * 16;
  for (int i = tid; i < 16 * 128; i += 256) {
    int row = node0 + (i >> 7);
    float v = H1[(size_t)(row < Nn ? row : Nn - 1) * 128 + (i & 127)];
    sH1f[i] = v;
    sH1b[i] = f2bf(v);
  }
  __syncthreads();
  int w = tid >> 5, lane = tid & 31, q = lane >> 4, mr = lane & 15;
  // GEMM A: 16x128 @ 128x512; A-fragments hoisted (shared by all 4 N-tiles per wave)
  {
    const __bf16* rowA = sH1b + mr * 128;
    v16bf a0 = load_afrag(rowA, 0, q);
    v16bf a1 = load_afrag(rowA, 32, q);
    v16bf a2 = load_afrag(rowA, 64, q);
    v16bf a3 = load_afrag(rowA, 96, q);
    for (int t = w; t < 32; t += 8) {
      int ncol = t * 16 + mr;
      v8f acc = {};
      const __bf16* Bt = W1t + (size_t)ncol * 128;
      v16bf b0 = *(const v16bf*)(Bt + 0 + q * 16);
      v16bf b1v = *(const v16bf*)(Bt + 32 + q * 16);
      v16bf b2v = *(const v16bf*)(Bt + 64 + q * 16);
      v16bf b3 = *(const v16bf*)(Bt + 96 + q * 16);
      acc = __builtin_amdgcn_wmma_f32_16x16x32_bf16(false, a0, false, b0, (short)0, acc, false, false);
      acc = __builtin_amdgcn_wmma_f32_16x16x32_bf16(false, a1, false, b1v, (short)0, acc, false, false);
      acc = __builtin_amdgcn_wmma_f32_16x16x32_bf16(false, a2, false, b2v, (short)0, acc, false, false);
      acc = __builtin_amdgcn_wmma_f32_16x16x32_bf16(false, a3, false, b3, (short)0, acc, false, false);
      float bias = b1[ncol];
      for (int r = 0; r < 8; ++r) {
        float v = acc[r] + bias;
        float g = 0.5f * v * (1.f + erff(v * 0.70710678118654752f));  // exact GELU
        sMid[(q * 8 + r) * 512 + ncol] = f2bf(g);
      }
    }
  }
  __syncthreads();
  // GEMM B: 16x512 @ 512x128, one 16x16 N-tile per wave
  {
    int ncol = w * 16 + mr;
    v8f acc = {};
    const __bf16* Bt = W2t + (size_t)ncol * 512;
    const __bf16* rowA = sMid + mr * 512;
    for (int kk = 0; kk < 512; kk += 32) {
      v16bf a = load_afrag(rowA, kk, q);
      v16bf b = *(const v16bf*)(Bt + kk + q * 16);
      acc = __builtin_amdgcn_wmma_f32_16x16x32_bf16(false, a, false, b, (short)0, acc, false, false);
    }
    float bias = b2[ncol];
    for (int r = 0; r < 8; ++r) sOut[(q * 8 + r) * 128 + ncol] = acc[r] + bias;
  }
  __syncthreads();
  // residual + LayerNorm over each of the 16 rows (16 threads per row)
  int rr = tid >> 4, j = tid & 15;
  float s = 0.f, qsum = 0.f;
  for (int c = j * 8; c < j * 8 + 8; ++c) {
    float v = sH1f[rr * 128 + c] + sOut[rr * 128 + c];
    s += v; qsum += v * v;
  }
  sRedS[tid] = s; sRedQ[tid] = qsum;
  __syncthreads();
  if (j == 0) {
    float S = 0.f, Q = 0.f;
    for (int k2 = 0; k2 < 16; ++k2) { S += sRedS[rr * 16 + k2]; Q += sRedQ[rr * 16 + k2]; }
    float mu = S * (1.f / 128.f);
    float var = Q * (1.f / 128.f) - mu * mu;
    sMu[rr] = mu;
    sRi[rr] = rsqrtf(var + 1e-5f);
  }
  __syncthreads();
  int grow = node0 + rr;
  if (grow < Nn) {
    float mu = sMu[rr], ri = sRi[rr];
    float4 o0, o1;
    int c0 = rr * 128 + j * 8;
    float v0 = sH1f[c0 + 0] + sOut[c0 + 0];
    float v1 = sH1f[c0 + 1] + sOut[c0 + 1];
    float v2 = sH1f[c0 + 2] + sOut[c0 + 2];
    float v3 = sH1f[c0 + 3] + sOut[c0 + 3];
    float v4 = sH1f[c0 + 4] + sOut[c0 + 4];
    float v5 = sH1f[c0 + 5] + sOut[c0 + 5];
    float v6 = sH1f[c0 + 6] + sOut[c0 + 6];
    float v7 = sH1f[c0 + 7] + sOut[c0 + 7];
    int cg = j * 8;
    o0.x = gamma[cg + 0] * (v0 - mu) * ri + beta[cg + 0];
    o0.y = gamma[cg + 1] * (v1 - mu) * ri + beta[cg + 1];
    o0.z = gamma[cg + 2] * (v2 - mu) * ri + beta[cg + 2];
    o0.w = gamma[cg + 3] * (v3 - mu) * ri + beta[cg + 3];
    o1.x = gamma[cg + 4] * (v4 - mu) * ri + beta[cg + 4];
    o1.y = gamma[cg + 5] * (v5 - mu) * ri + beta[cg + 5];
    o1.z = gamma[cg + 6] * (v6 - mu) * ri + beta[cg + 6];
    o1.w = gamma[cg + 7] * (v7 - mu) * ri + beta[cg + 7];
    float* dst = OUT + (size_t)grow * 128 + cg;
    *(float4*)dst = o0;
    *(float4*)(dst + 4) = o1;
  }
}

// ---------- host launcher ----------
extern "C" void kernel_launch(void* const* d_in, const int* in_sizes, int n_in,
                              void* d_out, int out_size, void* d_ws, size_t ws_size,
                              hipStream_t stream) {
  const float* x        = (const float*)d_in[0];
  const long long* ei   = (const long long*)d_in[1];
  const float* W_gat    = (const float*)d_in[2];
  const float* att_src  = (const float*)d_in[3];
  const float* att_dst  = (const float*)d_in[4];
  const float* b_gat    = (const float*)d_in[5];
  const float* gamma    = (const float*)d_in[6];
  const float* beta     = (const float*)d_in[7];
  const float* W1       = (const float*)d_in[8];
  const float* b1       = (const float*)d_in[9];
  const float* W2       = (const float*)d_in[10];
  const float* b2       = (const float*)d_in[11];
  float* out = (float*)d_out;
  int Nn = in_sizes[0] / 128;
  int Ee = in_sizes[1] / 2;

  // workspace carve-out (~167 MB)
  float* h       = (float*)d_ws;
  float* a_src   = h + (size_t)Nn * 128;
  float* a_dst   = a_src + (size_t)Nn * 8;
  unsigned* smax = (unsigned*)(a_dst + (size_t)Nn * 8);
  float* denom   = (float*)(smax + (size_t)Nn * 8);
  float* agg     = denom + (size_t)Nn * 8;
  float* h1      = agg + (size_t)Nn * 128;
  __bf16* Wgt    = (__bf16*)(h1 + (size_t)Nn * 128);
  __bf16* W1t    = Wgt + 128 * 128;
  __bf16* W2t    = W1t + 128 * 512;

  unsigned gW = (128 * 128 + 2 * 128 * 512 + 255) / 256;
  k_wconv<<<gW, 256, 0, stream>>>(W_gat, W1, W2, Wgt, W1t, W2t);

  size_t initN = (size_t)Nn * (8 + 8 + 128);
  k_init<<<(unsigned)((initN + 255) / 256), 256, 0, stream>>>(smax, denom, agg, Nn);

  k_gemm_gat<<<(unsigned)((Nn + 15) / 16), 256, 0, stream>>>(x, Wgt, h, Nn);

  k_att<<<(unsigned)(((size_t)Nn * 8 + 255) / 256), 256, 0, stream>>>(h, att_src, att_dst, a_src, a_dst, Nn);

  int EN = Ee + Nn;
  k_edge_max<<<(unsigned)((EN + 255) / 256), 256, 0, stream>>>(ei, a_src, a_dst, smax, Ee, Nn);
  k_edge_sum<<<(unsigned)((EN + 255) / 256), 256, 0, stream>>>(ei, a_src, a_dst, smax, denom, Ee, Nn);
  k_edge_scatter<<<(unsigned)(((size_t)EN * 8 + 255) / 256), 256, 0, stream>>>(ei, a_src, a_dst, smax, denom, h, agg, Ee, Nn);

  k_ln1<<<(unsigned)((Nn + 7) / 8), 256, 0, stream>>>(x, agg, b_gat, gamma, beta, h1, Nn);

  k_ffn<<<(unsigned)((Nn + 15) / 16), 256, 0, stream>>>(h1, W1t, b1, W2t, b2, gamma, beta, out, Nn);
}
